// MultiHeadAttention_3143916061087
// MI455X (gfx1250) — matrix operand order repaired and checked by reading, not compiled, not run on hardware
//
#include <hip/hip_runtime.h>
#include <math.h>

#define BB 4
#define TT 2048
#define CC 384
#define HH 6
#define DD 64
#define MM (BB*TT)   // 8192

typedef __attribute__((ext_vector_type(16))) _Float16 v16h;
typedef __attribute__((ext_vector_type(8)))  _Float16 v8h;
typedef __attribute__((ext_vector_type(4)))  _Float16 v4h;
typedef __attribute__((ext_vector_type(8)))  float    v8f;

// Build a 16x32 f16 A-fragment: lane l holds row (l&15),
// K = {kh..kh+15} in elems 0..15, kh=(l>>4)*16 (contiguous order).
// p must point at (row, kbase) of a row-major f16 matrix slice.
__device__ __forceinline__ v16h make_afrag(const _Float16* p) {
  return *(const v16h*)p;
}

// ---------- conversion kernels ----------
__global__ void cvt_f32_f16(const float* __restrict__ in, _Float16* __restrict__ out, int n4) {
  int i = blockIdx.x * blockDim.x + threadIdx.x;
  if (i < n4) {
    float4 v = ((const float4*)in)[i];
    v4h h; h[0] = (_Float16)v.x; h[1] = (_Float16)v.y; h[2] = (_Float16)v.z; h[3] = (_Float16)v.w;
    *(v4h*)(out + 4 * i) = h;
  }
}

// W: [K=384][N=384] row-major f32  ->  Wt: [N][K] f16
__global__ void wconv_t(const float* __restrict__ W, _Float16* __restrict__ Wt) {
  int idx = blockIdx.x * blockDim.x + threadIdx.x;
  if (idx < CC * CC) {
    int n = idx / CC, k = idx % CC;
    Wt[idx] = (_Float16)W[k * CC + n];
  }
}

// ---------- WMMA GEMM:  out[M,384] = A[M,384] @ Bt[384,384]^T + bias ----------
// MODE 0: f32 row-major out; MODE 1: f16 row-major out; MODE 2: f16 [b,h,d,t] out (for V)
template <int MODE>
__global__ __launch_bounds__(256) void gemm_wmma(const _Float16* __restrict__ A,
                                                 const _Float16* __restrict__ Bt,
                                                 const float* __restrict__ bias,
                                                 void* __restrict__ out) {
  const int lane = threadIdx.x & 31;
  const int wave = threadIdx.x >> 5;
  const int m0 = blockIdx.x * 128 + wave * 16;
  const int n0 = blockIdx.y * 64;
  const int ar = lane & 15;
  const int kb = (lane >> 4) << 3;  // 0 / 8
  const int kh = (lane >> 4) << 4;  // 0 / 16

  v8f acc[4] = {};
  const _Float16* arow = A + (size_t)(m0 + ar) * CC;

  for (int kk = 0; kk < CC; kk += 32) {
    __builtin_prefetch(arow + kk + 64, 0, 0);     // global_prefetch_b8
    v16h a = make_afrag(arow + kk + kh);
#pragma unroll
    for (int j = 0; j < 4; ++j) {
      v16h b = *(const v16h*)(Bt + (size_t)(n0 + j * 16 + ar) * CC + kk + kh);
      acc[j] = __builtin_amdgcn_wmma_f32_16x16x32_f16(false, a, false, b,
                                                      (short)0, acc[j], false, false);
    }
  }

#pragma unroll
  for (int j = 0; j < 4; ++j) {
#pragma unroll
    for (int i = 0; i < 8; ++i) {
      int row = m0 + i + ((lane >> 4) << 3);
      int col = n0 + j * 16 + (lane & 15);
      float v = acc[j][i] + bias[col];
      if (MODE == 0) {
        ((float*)out)[(size_t)row * CC + col] = v;
      } else if (MODE == 1) {
        ((_Float16*)out)[(size_t)row * CC + col] = (_Float16)v;
      } else {
        int b = row / TT, t = row % TT;
        int h = col >> 6, d = col & 63;
        ((_Float16*)out)[(((size_t)(b * HH + h)) * DD + d) * TT + t] = (_Float16)v;
      }
    }
  }
}

// ---------- flash attention: one wave = 16 query rows of one (b,h) ----------
__global__ __launch_bounds__(128) void attn_wmma(const _Float16* __restrict__ Q,
                                                 const _Float16* __restrict__ K,
                                                 const _Float16* __restrict__ Vt,
                                                 _Float16* __restrict__ CTX) {
  __shared__ _Float16 lds_p[4 * 16 * 32];  // per-wave 16x32 f16 P tile
  const int lane = threadIdx.x & 31;
  const int wave = threadIdx.x >> 5;
  const int b = blockIdx.y / HH;
  const int h = blockIdx.y % HH;
  const int t0 = blockIdx.x * 64 + wave * 16;
  const int ar = lane & 15;
  const int kb = (lane >> 4) << 3;
  const int kh = (lane >> 4) << 4;
  const int rhalf = (lane >> 4) << 3;

  const _Float16* Qb = Q + (size_t)(b * TT) * CC + h * DD;
  const _Float16* Kb = K + (size_t)(b * TT) * CC + h * DD;
  const _Float16* Vb = Vt + ((size_t)(b * HH + h)) * DD * TT;
  _Float16* Pl = lds_p + wave * 512;

  const v16h qa0 = make_afrag(Qb + (size_t)(t0 + ar) * CC + kh);
  const v16h qa1 = make_afrag(Qb + (size_t)(t0 + ar) * CC + 32 + kh);

  v8f o[4] = {};
  float mrow[8], lrow[8];
#pragma unroll
  for (int i = 0; i < 8; ++i) { mrow[i] = -INFINITY; lrow[i] = 0.f; }

  const int tmax = t0 + 15;
  const float sc = 0.125f;  // 1/sqrt(64), folded into exp

  for (int s0 = 0; s0 <= tmax; s0 += 32) {
    // S = Q . K^T for key cols [s0..s0+31]
    v8f sa0 = {}, sa1 = {};
#pragma unroll
    for (int kd = 0; kd < 2; ++kd) {
      v16h kf0 = *(const v16h*)(Kb + (size_t)(s0 + ar) * CC + kd * 32 + kh);
      v16h kf1 = *(const v16h*)(Kb + (size_t)(s0 + 16 + ar) * CC + kd * 32 + kh);
      v16h qa = (kd == 0) ? qa0 : qa1;
      sa0 = __builtin_amdgcn_wmma_f32_16x16x32_f16(false, qa, false, kf0, (short)0, sa0, false, false);
      sa1 = __builtin_amdgcn_wmma_f32_16x16x32_f16(false, qa, false, kf1, (short)0, sa1, false, false);
    }
    // online softmax, C-layout: VGPR i -> row t0+i+rhalf, lane&15 -> col
#pragma unroll
    for (int i = 0; i < 8; ++i) {
      int r = t0 + i + rhalf;
      int c0 = s0 + (lane & 15);
      float e0 = (c0 <= r) ? sa0[i] : -INFINITY;
      float e1 = (c0 + 16 <= r) ? sa1[i] : -INFINITY;
      float mx = fmaxf(e0, e1);
#pragma unroll
      for (int msk = 1; msk <= 8; msk <<= 1) mx = fmaxf(mx, __shfl_xor(mx, msk));
      float mnew = fmaxf(mrow[i], mx);
      float corr = __expf((mrow[i] - mnew) * sc);
      mrow[i] = mnew;
      lrow[i] *= corr;
#pragma unroll
      for (int j = 0; j < 4; ++j) o[j][i] *= corr;
      float p0 = __expf((e0 - mnew) * sc);
      float p1 = __expf((e1 - mnew) * sc);
      lrow[i] += p0 + p1;
      int rl = i + rhalf;
      Pl[rl * 32 + (lane & 15)] = (_Float16)p0;
      Pl[rl * 32 + 16 + (lane & 15)] = (_Float16)p1;
    }
    asm volatile("s_wait_dscnt 0" ::: "memory");
    // re-read P in A-fragment layout, accumulate O += P . V
    v16h pa = make_afrag(Pl + ar * 32 + kh);
#pragma unroll
    for (int j = 0; j < 4; ++j) {
      v16h vb = *(const v16h*)(Vb + (size_t)(j * 16 + ar) * TT + s0 + kh);
      o[j] = __builtin_amdgcn_wmma_f32_16x16x32_f16(false, pa, false, vb, (short)0, o[j], false, false);
    }
  }

  // normalize rows and store ctx (f16, row-major [b*T, h*64+d])
#pragma unroll
  for (int i = 0; i < 8; ++i) {
    float l = lrow[i];
#pragma unroll
    for (int msk = 1; msk <= 8; msk <<= 1) l += __shfl_xor(l, msk);
    float inv = 1.0f / l;
    int r = t0 + i + rhalf;
#pragma unroll
    for (int j = 0; j < 4; ++j)
      CTX[(size_t)(b * TT + r) * CC + h * DD + j * 16 + (lane & 15)] = (_Float16)(o[j][i] * inv);
  }
}

extern "C" void kernel_launch(void* const* d_in, const int* in_sizes, int n_in,
                              void* d_out, int out_size, void* d_ws, size_t ws_size,
                              hipStream_t stream) {
  (void)in_sizes; (void)n_in; (void)out_size; (void)ws_size;
  const float* x  = (const float*)d_in[0];
  const float* Wq = (const float*)d_in[1];
  const float* bq = (const float*)d_in[2];
  const float* Wk = (const float*)d_in[3];
  const float* bk = (const float*)d_in[4];
  const float* Wv = (const float*)d_in[5];
  const float* bv = (const float*)d_in[6];
  const float* Wo = (const float*)d_in[7];
  const float* bo = (const float*)d_in[8];

  const size_t MC = (size_t)MM * CC;  // 3,145,728 halfs
  const size_t WW = (size_t)CC * CC;  // 147,456 halfs
  _Float16* xh   = (_Float16*)d_ws;
  _Float16* Qh   = xh + MC;
  _Float16* Kh   = Qh + MC;
  _Float16* Vt   = Kh + MC;   // [b,h,d,t]
  _Float16* CTXh = Vt + MC;
  _Float16* WqT  = CTXh + MC;
  _Float16* WkT  = WqT + WW;
  _Float16* WvT  = WkT + WW;
  _Float16* WoT  = WvT + WW;

  cvt_f32_f16<<<dim3((unsigned)(MC / 4 / 256)), 256, 0, stream>>>(x, xh, (int)(MC / 4));
  dim3 wgrid((unsigned)((CC * CC + 255) / 256));
  wconv_t<<<wgrid, 256, 0, stream>>>(Wq, WqT);
  wconv_t<<<wgrid, 256, 0, stream>>>(Wk, WkT);
  wconv_t<<<wgrid, 256, 0, stream>>>(Wv, WvT);
  wconv_t<<<wgrid, 256, 0, stream>>>(Wo, WoT);

  dim3 ggrid(MM / 128, CC / 64);  // (64, 6)
  gemm_wmma<1><<<ggrid, 256, 0, stream>>>(xh, WqT, bq, Qh);
  gemm_wmma<1><<<ggrid, 256, 0, stream>>>(xh, WkT, bk, Kh);
  gemm_wmma<2><<<ggrid, 256, 0, stream>>>(xh, WvT, bv, Vt);

  attn_wmma<<<dim3(TT / 64, BB * HH), 128, 0, stream>>>(Qh, Kh, Vt, CTXh);

  gemm_wmma<0><<<ggrid, 256, 0, stream>>>(CTXh, WoT, bo, d_out);
}
